// MultiHeadAttention_7206955123309
// MI455X (gfx1250) — matrix operand order repaired and checked by reading, not compiled, not run on hardware
//
#include <hip/hip_runtime.h>
#include <hip/hip_bf16.h>

typedef __attribute__((ext_vector_type(16))) _Float16 v16h;
typedef __attribute__((ext_vector_type(8)))  _Float16 v8h;
typedef __attribute__((ext_vector_type(8)))  float    v8f;
typedef unsigned int __attribute__((ext_vector_type(4))) v4u;
typedef int __attribute__((ext_vector_type(8))) v8i_;
typedef int __attribute__((ext_vector_type(4))) v4i_;

#define B_  8
#define S_  1024
#define D_  1024
#define H_  16
#define DK_ 64

// ---------- helpers ----------

static __device__ __forceinline__ v16h cat8(v8h a, v8h b) {
  v16h r;
  #pragma unroll
  for (int i = 0; i < 8; ++i) { r[i] = a[i]; r[i + 8] = b[i]; }
  return r;
}

template <int XMASK>
static __device__ __forceinline__ float swz_xor(float x) {
  // ds_swizzle_b32 group-of-32 XOR pattern: offset = {xor<<10 | or<<5 | and}
  int i = __builtin_amdgcn_ds_swizzle(__float_as_int(x), (XMASK << 10) | 0x1f);
  return __int_as_float(i);
}

static __device__ __forceinline__ v8f wmma16(v16h a, v16h b, v8f c) {
  return __builtin_amdgcn_wmma_f32_16x16x32_f16(
      /*neg_a=*/false, a, /*neg_b=*/false, b,
      /*c_mod=*/(short)0, c, /*reuse_a=*/false, /*reuse_b=*/false);
}

// ---------- tiled GEMM: C = A @ W + bias (+residual) ----------
// MODE 0: A is f32 (converted to f16 while staging), out f16  = A@W + bias
// MODE 1: A is f16 (TDM tensor_load_to_lds staging), out f32 = A@W + bias + res
// Block: 256 threads (8 waves), tile 128x128, k-step 32.
// Per wave: 32x64 of C = 2x4 grid of 16x16 WMMA tiles.

template <int MODE>
__global__ __launch_bounds__(256) void gemm_kernel(
    const float* __restrict__ A32, const _Float16* __restrict__ A16,
    const float* __restrict__ W, const float* __restrict__ bias,
    const float* __restrict__ res, _Float16* __restrict__ out16,
    float* __restrict__ out32, int M, int N, int K) {
  __shared__ _Float16 lds_a[128 * 40];   // A tile 128x32, padded stride 40
  __shared__ _Float16 lds_bt[128 * 40];  // W tile transposed: [n][k], stride 40

  const int tid  = threadIdx.x;
  const int wave = tid >> 5, lane = tid & 31;
  const int hi = lane >> 4, lo = lane & 15;
  const int waveM = wave >> 1, waveN = wave & 1;
  const int m0 = blockIdx.y * 128, n0 = blockIdx.x * 128;

  v8f zero = {};
  v8f acc[2][4];
  #pragma unroll
  for (int mt = 0; mt < 2; ++mt)
    #pragma unroll
    for (int nt = 0; nt < 4; ++nt) acc[mt][nt] = zero;

  for (int k0 = 0; k0 < K; k0 += 32) {
    __syncthreads();
    // ---- stage A tile ----
    if (MODE == 0) {
      // f32 -> f16 conversion while staging (VALU path; TDM cannot convert)
      #pragma unroll
      for (int i = 0; i < 4; ++i) {
        int idx = i * 256 + tid;          // 0..1023 float4's
        int row = idx >> 3, c4 = idx & 7;
        const float4 w =
            *(const float4*)(A32 + (size_t)(m0 + row) * K + k0 + c4 * 4);
        _Float16* dst = lds_a + row * 40 + c4 * 4;
        dst[0] = (_Float16)w.x; dst[1] = (_Float16)w.y;
        dst[2] = (_Float16)w.z; dst[3] = (_Float16)w.w;
      }
      if (k0 + 32 < K)
        __builtin_prefetch(A32 + (size_t)(m0 + (tid >> 1)) * K + k0 + 32, 0, 1);
    } else {
#if __has_builtin(__builtin_amdgcn_tensor_load_to_lds) && \
    __has_builtin(__builtin_amdgcn_s_wait_tensorcnt)
      // Raw f16 tile: stage with the Tensor Data Mover (6-arg clang-23 form).
      // D# group0: count=1 | lds_addr | global_addr(57b) | type=2
      // D# group1: data_size=2B, pad_enable, pad_interval=16dw, pad_amount=4dw
      //            (reproduces the 40-half LDS row stride), tile 32x128,
      //            tensor_dim0_stride = K.
      if (wave == 0) {
        unsigned lds_base = (unsigned)(size_t)(&lds_a[0]);
        unsigned long long ga =
            (unsigned long long)(const void*)(A16 + (size_t)m0 * K + k0);
        v4u g0;
        g0[0] = 1u;                                  // count=1 (valid)
        g0[1] = lds_base;                            // lds_addr
        g0[2] = (unsigned)(ga & 0xffffffffu);        // global_addr[31:0]
        g0[3] = (unsigned)((ga >> 32) & 0x01ffffffu) // global_addr[56:32]
                | (2u << 30);                        // type=2 ("image")
        v8i_ g1;
        g1[0] = (int)((1u << 16)    // data_size = 2 bytes
                    | (1u << 20)    // pad_enable
                    | (3u << 22)    // pad_interval: 16 DWORDs
                    | (3u << 25));  // pad_amount: 4 DWORDs
        g1[1] = (int)(((unsigned)K & 0xffffu) << 16);      // tensor_dim0 lo16
        g1[2] = (int)((((unsigned)K >> 16) & 0xffffu)      // tensor_dim0 hi16
                    | (((unsigned)M & 0xffffu) << 16));    // tensor_dim1 lo16
        g1[3] = (int)((((unsigned)M >> 16) & 0xffffu)      // tensor_dim1 hi16
                    | (32u << 16));                        // tile_dim0 = 32
        g1[4] = (int)128u;                                 // tile_dim1 = 128
        g1[5] = (int)(unsigned)K;                          // dim0_stride[31:0]
        g1[6] = 0;                                         // stride hi / dim1_stride
        g1[7] = 0;
        v4i_ gz4 = {0, 0, 0, 0};
        v8i_ gz8 = {0, 0, 0, 0, 0, 0, 0, 0};
        __builtin_amdgcn_tensor_load_to_lds(g0, g1, gz4, gz4, gz8, 0);
        __builtin_amdgcn_s_wait_tensorcnt(0);
      }
#else
      #pragma unroll
      for (int i = 0; i < 2; ++i) {
        int idx = i * 256 + tid;          // 0..511 v8h's
        int row = idx >> 2, c8 = idx & 3;
        v8h v = *(const v8h*)(A16 + (size_t)(m0 + row) * K + k0 + c8 * 8);
        *(v8h*)(lds_a + row * 40 + c8 * 8) = v;
      }
#endif
      if (k0 + 32 < K)
        __builtin_prefetch(A16 + (size_t)(m0 + (tid >> 1)) * K + k0 + 32, 0, 1);
    }
    // ---- stage W tile transposed: lds_bt[n][k] = W[k0+k][n0+n] ----
    #pragma unroll
    for (int i = 0; i < 4; ++i) {
      int idx = i * 256 + tid;            // 0..1023 float4's
      int kr = idx >> 5, nc4 = idx & 31;
      const float4 w =
          *(const float4*)(W + (size_t)(k0 + kr) * N + n0 + nc4 * 4);
      lds_bt[(nc4 * 4 + 0) * 40 + kr] = (_Float16)w.x;
      lds_bt[(nc4 * 4 + 1) * 40 + kr] = (_Float16)w.y;
      lds_bt[(nc4 * 4 + 2) * 40 + kr] = (_Float16)w.z;
      lds_bt[(nc4 * 4 + 3) * 40 + kr] = (_Float16)w.w;
    }
    if (k0 + 32 < K)
      __builtin_prefetch(W + (size_t)(k0 + 32 + (tid >> 3)) * N + n0, 0, 1);
    __syncthreads();

    // ---- wave-level WMMA over the 32-wide k slab ----
    v16h afrag[2];
    #pragma unroll
    for (int mt = 0; mt < 2; ++mt) {
      const _Float16* p = lds_a + (waveM * 32 + mt * 16 + lo) * 40 + 8 * hi;
      afrag[mt] = cat8(*(const v8h*)p, *(const v8h*)(p + 16));
    }
    #pragma unroll
    for (int nt = 0; nt < 4; ++nt) {
      const _Float16* p = lds_bt + (waveN * 64 + nt * 16 + lo) * 40 + 8 * hi;
      v16h bfrag = cat8(*(const v8h*)p, *(const v8h*)(p + 16));
      #pragma unroll
      for (int mt = 0; mt < 2; ++mt)
        acc[mt][nt] = wmma16(afrag[mt], bfrag, acc[mt][nt]);
    }
  }

  // ---- epilogue ----
  #pragma unroll
  for (int nt = 0; nt < 4; ++nt) {
    int gn = n0 + waveN * 64 + nt * 16 + lo;
    float bv = bias[gn];
    #pragma unroll
    for (int mt = 0; mt < 2; ++mt) {
      int gmb = m0 + waveM * 32 + mt * 16 + hi * 8;
      #pragma unroll
      for (int r = 0; r < 8; ++r) {
        size_t o = (size_t)(gmb + r) * N + gn;
        float v = acc[mt][nt][r] + bv;
        if (MODE == 0) out16[o] = (_Float16)v;
        else           out32[o] = v + res[o];
      }
    }
  }
}

// ---------- flash attention: per (b,h), 64 queries/block, online softmax ----
// Block: 128 threads = 4 waves; each wave owns 16 query rows.
// Key blocks of 32: QK^T via WMMA (scores C layout), mask + online softmax,
// P transposed to A layout via wave-private LDS, O += P@V via WMMA with V
// staged transposed in LDS.

__global__ __launch_bounds__(128) void attn_kernel(
    const _Float16* __restrict__ qh, const _Float16* __restrict__ kh,
    const _Float16* __restrict__ vh, const int* __restrict__ mask,
    _Float16* __restrict__ outh) {
  __shared__ _Float16 lds_vt[64 * 40];      // V^T: [dk][key], stride 40
  __shared__ _Float16 lds_p[4 * 16 * 40];   // per-wave P scratch 16x32

  const int tid  = threadIdx.x;
  const int wave = tid >> 5, lane = tid & 31;
  const int hi = lane >> 4, lo = lane & 15;
  const int b = blockIdx.y / H_, h = blockIdx.y % H_;
  const int q0 = blockIdx.x * 64;
  const int qrow = q0 + wave * 16 + lo;

  // Q fragments: 16x64 = two 16x32 A-fragments, loaded once
  const _Float16* qp = qh + ((size_t)(b * S_ + qrow)) * D_ + h * DK_;
  v16h qf[2];
  #pragma unroll
  for (int j = 0; j < 2; ++j)
    qf[j] = cat8(*(const v8h*)(qp + 32 * j + 8 * hi),
                 *(const v8h*)(qp + 32 * j + 16 + 8 * hi));

  v8f zero = {};
  float mrow[8], lrow[8];
  v8f o[4];
  #pragma unroll
  for (int r = 0; r < 8; ++r) { mrow[r] = -1e30f; lrow[r] = 0.0f; }
  #pragma unroll
  for (int t = 0; t < 4; ++t) o[t] = zero;

  _Float16* pbuf = lds_p + wave * 16 * 40;

  for (int kb = 0; kb < S_; kb += 32) {
    __syncthreads();  // all waves done reading previous V^T
    // ---- stage V^T tile: lds_vt[dk][key] ----
    #pragma unroll
    for (int i = 0; i < 2; ++i) {
      int idx = i * 128 + tid;            // 0..255 v8h's
      int key = idx >> 3, c8 = idx & 7;
      v8h v = *(const v8h*)(vh + ((size_t)(b * S_ + kb + key)) * D_ +
                            h * DK_ + c8 * 8);
      #pragma unroll
      for (int e = 0; e < 8; ++e) lds_vt[(c8 * 8 + e) * 40 + key] = v[e];
    }
    __syncthreads();

    // ---- scores S = Q @ K^T (two 16x16 tiles over 32 keys) ----
    v8f sc[2];
    int mk[2];
    #pragma unroll
    for (int nt = 0; nt < 2; ++nt) {
      v8f s = zero;
      int key = kb + nt * 16 + lo;
      const _Float16* kp =
          kh + ((size_t)(b * S_ + key)) * D_ + h * DK_ + 8 * hi;
      #pragma unroll
      for (int j = 0; j < 2; ++j) {
        v16h kf = cat8(*(const v8h*)(kp + 32 * j), *(const v8h*)(kp + 32 * j + 16));
        s = wmma16(qf[j], kf, s);
      }
      mk[nt] = mask[b * S_ + key];        // key mask: same N for all 8 rows
      #pragma unroll
      for (int r = 0; r < 8; ++r)
        s[r] = mk[nt] ? s[r] * 0.125f : -1e20f;
      sc[nt] = s;
    }

    // ---- online softmax (row reductions across 16 lanes via ds_swizzle) ----
    float nm[8], corr[8];
    #pragma unroll
    for (int r = 0; r < 8; ++r) {
      float t = fmaxf(sc[0][r], sc[1][r]);
      t = fmaxf(t, swz_xor<1>(t)); t = fmaxf(t, swz_xor<2>(t));
      t = fmaxf(t, swz_xor<4>(t)); t = fmaxf(t, swz_xor<8>(t));
      nm[r] = fmaxf(mrow[r], t);
      corr[r] = __expf(mrow[r] - nm[r]);
      mrow[r] = nm[r];
    }
    #pragma unroll
    for (int nt = 0; nt < 2; ++nt)
      #pragma unroll
      for (int r = 0; r < 8; ++r)
        sc[nt][r] = __expf(sc[nt][r] - nm[r]);
    #pragma unroll
    for (int r = 0; r < 8; ++r) {
      float t = sc[0][r] + sc[1][r];
      t += swz_xor<1>(t); t += swz_xor<2>(t);
      t += swz_xor<4>(t); t += swz_xor<8>(t);
      lrow[r] = lrow[r] * corr[r] + t;
    }
    #pragma unroll
    for (int t4 = 0; t4 < 4; ++t4)
      #pragma unroll
      for (int r = 0; r < 8; ++r) o[t4][r] *= corr[r];

    // ---- P: C layout -> A layout via wave-private LDS ----
    #pragma unroll
    for (int nt = 0; nt < 2; ++nt)
      #pragma unroll
      for (int r = 0; r < 8; ++r)
        pbuf[(r + 8 * hi) * 40 + nt * 16 + lo] = (_Float16)sc[nt][r];
    __syncthreads();

    const _Float16* pp = pbuf + lo * 40 + 8 * hi;
    v16h pf = cat8(*(const v8h*)pp, *(const v8h*)(pp + 16));
    #pragma unroll
    for (int t4 = 0; t4 < 4; ++t4) {
      const _Float16* vp = lds_vt + (t4 * 16 + lo) * 40 + 8 * hi;
      v16h vf = cat8(*(const v8h*)vp, *(const v8h*)(vp + 16));
      o[t4] = wmma16(pf, vf, o[t4]);
    }
  }

  // ---- normalize + store (f16, (B,S,D) layout) ----
  float rl[8];
  #pragma unroll
  for (int r = 0; r < 8; ++r) rl[r] = 1.0f / lrow[r];
  #pragma unroll
  for (int t4 = 0; t4 < 4; ++t4) {
    int col = h * DK_ + t4 * 16 + lo;
    #pragma unroll
    for (int r = 0; r < 8; ++r) {
      int row = q0 + wave * 16 + 8 * hi + r;
      outh[((size_t)(b * S_ + row)) * D_ + col] = (_Float16)(o[t4][r] * rl[r]);
    }
  }
}

// ---------- LayerNorm over D=1024, one block per row ----------

__global__ __launch_bounds__(256) void ln_kernel(
    const float* __restrict__ x, const float* __restrict__ gamma,
    const float* __restrict__ beta, float* __restrict__ out) {
  __shared__ float ssum[8], ssq[8];
  const int row = blockIdx.x, tid = threadIdx.x;
  const float* xr = x + (size_t)row * D_;
  float v[4], s = 0.0f, q = 0.0f;
  #pragma unroll
  for (int i = 0; i < 4; ++i) {
    v[i] = xr[i * 256 + tid];
    s += v[i];
    q += v[i] * v[i];
  }
  s += swz_xor<1>(s); s += swz_xor<2>(s); s += swz_xor<4>(s);
  s += swz_xor<8>(s); s += swz_xor<16>(s);
  q += swz_xor<1>(q); q += swz_xor<2>(q); q += swz_xor<4>(q);
  q += swz_xor<8>(q); q += swz_xor<16>(q);
  if ((tid & 31) == 0) { ssum[tid >> 5] = s; ssq[tid >> 5] = q; }
  __syncthreads();
  float ts = 0.0f, tq = 0.0f;
  #pragma unroll
  for (int i = 0; i < 8; ++i) { ts += ssum[i]; tq += ssq[i]; }
  float mu = ts * (1.0f / D_);
  float var = tq * (1.0f / D_) - mu * mu;
  float rstd = rsqrtf(var + 1e-5f);
  #pragma unroll
  for (int i = 0; i < 4; ++i) {
    int c = i * 256 + tid;
    out[(size_t)row * D_ + c] = (v[i] - mu) * rstd * gamma[c] + beta[c];
  }
}

// ---------- launch ----------

extern "C" void kernel_launch(void* const* d_in, const int* in_sizes, int n_in,
                              void* d_out, int out_size, void* d_ws,
                              size_t ws_size, hipStream_t stream) {
  (void)in_sizes; (void)n_in; (void)out_size; (void)ws_size;
  const float* x1    = (const float*)d_in[0];
  const float* x2    = (const float*)d_in[1];
  const float* x3    = (const float*)d_in[2];
  const int*   mask  = (const int*)d_in[3];
  const float* Wq    = (const float*)d_in[4];
  const float* bq    = (const float*)d_in[5];
  const float* Wk    = (const float*)d_in[6];
  const float* bk    = (const float*)d_in[7];
  const float* Wo    = (const float*)d_in[8];
  const float* bo    = (const float*)d_in[9];
  const float* gamma = (const float*)d_in[10];
  const float* beta  = (const float*)d_in[11];

  char* ws = (char*)d_ws;
  const size_t SZH = (size_t)B_ * S_ * D_ * sizeof(_Float16);  // 16 MB
  _Float16* qh  = (_Float16*)(ws);
  _Float16* kh  = (_Float16*)(ws + SZH);
  _Float16* vh  = (_Float16*)(ws + 2 * SZH);
  _Float16* aoh = (_Float16*)(ws + 3 * SZH);
  float*    xb  = (float*)(ws);  // overlays qh/kh (dead after attention)

  dim3 gblk(256), ggrid(D_ / 128, (B_ * S_) / 128);
  gemm_kernel<0><<<ggrid, gblk, 0, stream>>>(x1, nullptr, Wq, bq, nullptr, qh,
                                             nullptr, B_ * S_, D_, D_);
  gemm_kernel<0><<<ggrid, gblk, 0, stream>>>(x2, nullptr, Wk, bk, nullptr, kh,
                                             nullptr, B_ * S_, D_, D_);
  gemm_kernel<0><<<ggrid, gblk, 0, stream>>>(x3, nullptr, Wk, bk, nullptr, vh,
                                             nullptr, B_ * S_, D_, D_);
  attn_kernel<<<dim3(S_ / 64, B_ * H_), dim3(128), 0, stream>>>(qh, kh, vh,
                                                                mask, aoh);
  gemm_kernel<1><<<ggrid, gblk, 0, stream>>>(nullptr, aoh, Wo, bo, x1, nullptr,
                                             xb, B_ * S_, D_, D_);
  ln_kernel<<<dim3(B_ * S_), dim3(256), 0, stream>>>(xb, gamma, beta,
                                                     (float*)d_out);
}
